// LinearMultiHeadAttention_32607391711939
// MI455X (gfx1250) — matrix operand order repaired and checked by reading, not compiled, not run on hardware
//
#include <hip/hip_runtime.h>
#include <hip/hip_bf16.h>

// ---------------------------------------------------------------------------
// Linear multi-head attention (ELU+1 feature map), MI455X / gfx1250, wave32.
// All GEMMs on v_wmma_f32_16x16x32_f16. Weights pre-transposed to f16 [N][K].
// Big GEMMs use a double-buffered software pipeline: async global->LDS copies
// (GLOBAL_LOAD_ASYNC_TO_LDS_B128, ASYNCcnt-tracked) for tile i+1 overlap the
// WMMA compute of tile i; one workgroup barrier per K-step.
// ---------------------------------------------------------------------------

typedef __attribute__((ext_vector_type(16))) _Float16 v16h;
typedef __attribute__((ext_vector_type(8)))  _Float16 v8h;
typedef __attribute__((ext_vector_type(4)))  _Float16 v4h;
typedef __attribute__((ext_vector_type(2)))  _Float16 v2h;
typedef __attribute__((ext_vector_type(8)))  float    v8f;
typedef __attribute__((ext_vector_type(4)))  int      v4i;

#define LMHA_B  4
#define LMHA_T  4096
#define LMHA_BT (LMHA_B * LMHA_T)       // 16384
#define LMHA_D  1024
#define LMHA_H  16
#define LMHA_DK 64
#define LMHA_DV 64
#define LMHA_EPS 1e-6f

// LDS row strides (halfs); multiples of 8 halfs (16B) keep v8h/b128 aligned.
#define STR32 40
#define STR64 72

#if __has_builtin(__builtin_amdgcn_global_load_async_to_lds_b128) && \
    __has_builtin(__builtin_amdgcn_s_wait_asynccnt)
#define HAS_ASYNC 1
#else
#define HAS_ASYNC 0
#endif

#define GLOBAL_AS __attribute__((address_space(1)))
#define LDS_AS    __attribute__((address_space(3)))

// Copy 16B global -> LDS. Async (ASYNCcnt-tracked) when the toolchain has it.
__device__ __forceinline__ void cp16(const _Float16* g, _Float16* l) {
#if HAS_ASYNC
    __builtin_amdgcn_global_load_async_to_lds_b128(
        (GLOBAL_AS v4i*)g, (LDS_AS v4i*)l, 0, 0);
#else
    *(v8h*)l = *(const v8h*)g;
#endif
}
__device__ __forceinline__ void wait_async() {
#if HAS_ASYNC
    __builtin_amdgcn_s_wait_asynccnt(0);
#endif
}

__device__ __forceinline__ v8f wmma_f16(v16h a, v16h b, v8f c) {
    return __builtin_amdgcn_wmma_f32_16x16x32_f16(
        false, a, false, b, (short)0, c, false, false);
}

// A fragment (16x32 f16): lane L holds row M=L%16; lanes 0-15: K={0..7,16..23},
// lanes 16-31: K={8..15,24..31}. LDS layout base[row*stride + k], k contiguous.
__device__ __forceinline__ v16h frag_a(const _Float16* base, int stride, int row, int lane) {
    const int kb = (lane & 16) ? 8 : 0;
    const v8h lo = *(const v8h*)(base + row * stride + kb);
    const v8h hi = *(const v8h*)(base + row * stride + kb + 16);
    v16h a;
#pragma unroll
    for (int i = 0; i < 8; ++i) { a[i] = lo[i]; a[i + 8] = hi[i]; }
    return a;
}

// B fragment (32x16 f16): lane L holds col N=L%16; lanes 0-15: K={0..7,16..23}, lanes 16-31: K={8..15,24..31}.
// LDS layout base[col*stride + k] (B stored transposed, [N][K]).
__device__ __forceinline__ v16h frag_b(const _Float16* base, int stride, int col, int lane) {
    const int kb = (lane & 16) ? 8 : 0;
    const v8h lo = *(const v8h*)(base + col * stride + kb);
    const v8h hi = *(const v8h*)(base + col * stride + kb + 16);
    v16h b;
#pragma unroll
    for (int i = 0; i < 8; ++i) { b[i] = lo[i]; b[i + 8] = hi[i]; }
    return b;
}

// ---------------------------------------------------------------------------
// Weight convert+transpose: W f32 [K][N] -> Wt f16 [N][K]. 64x64 tiles via LDS.
// ---------------------------------------------------------------------------
__global__ __launch_bounds__(256) void wconv_kernel(
    const float* __restrict__ W, _Float16* __restrict__ Wt)
{
    const int k0 = blockIdx.x * 64;
    const int n0 = blockIdx.y * 64;
    const int tid = threadIdx.x;
    __shared__ _Float16 ls[64 * STR64];    // [k][n]
#pragma unroll
    for (int j = 0; j < 4; ++j) {
        const int flat = tid + j * 256;    // 1024 float4 chunks
        const int kr = flat >> 4;
        const int nc = (flat & 15) * 4;
        const float4 v = *(const float4*)(W + (size_t)(k0 + kr) * LMHA_D + n0 + nc);
        v4h h = { (_Float16)v.x, (_Float16)v.y, (_Float16)v.z, (_Float16)v.w };
        *(v4h*)(ls + kr * STR64 + nc) = h;
    }
    __syncthreads();
#pragma unroll
    for (int j = 0; j < 2; ++j) {
        const int flat = tid + j * 256;    // 512 chunks of 8 halfs
        const int n  = flat >> 3;
        const int kg = (flat & 7) * 8;
        v8h o;
#pragma unroll
        for (int i = 0; i < 8; ++i) o[i] = ls[(kg + i) * STR64 + n];
        *(v8h*)(Wt + (size_t)(n0 + n) * LMHA_D + k0 + kg) = o;
    }
}

// ---------------------------------------------------------------------------
// GEMM: Y = act(A @ Wt^T + bias). A: [M x Kd] f32 or f16, Wt: f16 [N][K],
// Y: [M x N] f16 or f32. Block tile 128x128, 8 waves of 32x64, K-step 32.
// Double-buffered LDS pipeline, one barrier per K-step.
// ---------------------------------------------------------------------------
template <bool A_F16, bool PHI, bool OUT_F16>
__global__ __launch_bounds__(256) void gemm_kernel(
    const void* __restrict__ Aptr, const _Float16* __restrict__ Wt,
    const float* __restrict__ bias, void* __restrict__ Yptr,
    int M, int N, int Kd)
{
    __shared__ _Float16 lsA[2][128 * STR32];   // [m][k]
    __shared__ _Float16 lsB[2][128 * STR32];   // [n][k]

    const int tid  = threadIdx.x;
    const int lane = tid & 31;
    const int w    = tid >> 5;
    const int gm0  = blockIdx.x * 128;
    const int gn0  = blockIdx.y * 128;
    const int wm0  = (w & 3) * 32;   // 4 waves along M
    const int wn0  = (w >> 2) * 64;  // 2 waves along N

    v8f acc[2][4] = {};

    // ---- tile stager ------------------------------------------------------
    auto stage = [&](int buf, int k0) {
        if (A_F16) {
            const _Float16* A = (const _Float16*)Aptr;
#pragma unroll
            for (int j = 0; j < 2; ++j) {
                const int flat = tid + j * 256;
                const int row  = flat >> 2;
                const int cg   = (flat & 3) * 8;
                cp16(A + (size_t)(gm0 + row) * Kd + k0 + cg,
                     lsA[buf] + row * STR32 + cg);
            }
        } else {
            const float* A = (const float*)Aptr;
#pragma unroll
            for (int j = 0; j < 4; ++j) {
                const int flat = tid + j * 256;
                const int row  = flat >> 3;
                const int cg   = (flat & 7) * 4;
                const float4 v = *(const float4*)(A + (size_t)(gm0 + row) * Kd + k0 + cg);
                v4h h = { (_Float16)v.x, (_Float16)v.y, (_Float16)v.z, (_Float16)v.w };
                *(v4h*)(lsA[buf] + row * STR32 + cg) = h;
            }
        }
#pragma unroll
        for (int j = 0; j < 2; ++j) {
            const int flat = tid + j * 256;
            const int n  = flat >> 2;
            const int cg = (flat & 3) * 8;
            cp16(Wt + (size_t)(gn0 + n) * Kd + k0 + cg,
                 lsB[buf] + n * STR32 + cg);
        }
    };

    const int nk = Kd >> 5;      // K-steps of 32
    stage(0, 0);

    for (int i = 0; i < nk; ++i) {
        const int cur = i & 1;
        wait_async();            // this wave's copies into buf `cur` done
        __syncthreads();         // everyone's copies done; prev compute done
        if (i + 1 < nk) stage(cur ^ 1, (i + 1) << 5);   // overlap with WMMA

        v16h af[2], bf[4];
#pragma unroll
        for (int t = 0; t < 2; ++t) af[t] = frag_a(lsA[cur], STR32, wm0 + t * 16, lane);
#pragma unroll
        for (int t = 0; t < 4; ++t) bf[t] = frag_b(lsB[cur], STR32, wn0 + t * 16, lane);
#pragma unroll
        for (int mi = 0; mi < 2; ++mi)
#pragma unroll
            for (int ni = 0; ni < 4; ++ni)
                acc[mi][ni] = wmma_f16(af[mi], bf[ni], acc[mi][ni]);
    }

    // epilogue: C layout: VGPR r -> M = r + (lane<16?0:8), N = lane%16
#pragma unroll
    for (int ni = 0; ni < 4; ++ni) {
        const int n  = gn0 + wn0 + ni * 16 + (lane & 15);
        const float bv = bias[n];
#pragma unroll
        for (int mi = 0; mi < 2; ++mi) {
#pragma unroll
            for (int r = 0; r < 8; ++r) {
                const int m = gm0 + wm0 + mi * 16 + r + ((lane & 16) ? 8 : 0);
                float v = acc[mi][ni][r] + bv;
                if (PHI) v = (v > 0.f) ? (v + 1.f) : __expf(v);
                if (OUT_F16) ((_Float16*)Yptr)[(size_t)m * N + n] = (_Float16)v;
                else         ((float*)Yptr)[(size_t)m * N + n] = v;
            }
        }
    }
}

// ---------------------------------------------------------------------------
// ksum[b,h,d] = sum_t K[b,t,h,d]
// ---------------------------------------------------------------------------
__global__ __launch_bounds__(256) void ksum_kernel(
    const _Float16* __restrict__ Kf, float* __restrict__ ksum)
{
    const int bh = blockIdx.x;
    const int b = bh >> 4, h = bh & 15;
    const int tid = threadIdx.x;
    const int d = tid & 63;
    const int slice = tid >> 6;
    __shared__ float red[256];
    float s = 0.f;
    const size_t base = ((size_t)b * LMHA_T) * LMHA_D + (size_t)h * LMHA_DK + d;
    for (int t = slice * 1024; t < (slice + 1) * 1024; ++t)
        s += (float)Kf[base + (size_t)t * LMHA_D];
    red[tid] = s;
    __syncthreads();
    if (tid < 64)
        ksum[bh * 64 + d] = red[tid] + red[tid + 64] + red[tid + 128] + red[tid + 192];
}

// ---------------------------------------------------------------------------
// kv[b,h,kk,d] = sum_t V[b,t,h,kk] * K[b,t,h,d]
// ---------------------------------------------------------------------------
__global__ __launch_bounds__(256) void kv_kernel(
    const _Float16* __restrict__ Kf, const _Float16* __restrict__ Vf,
    float* __restrict__ kv)
{
    const int bh = blockIdx.x;
    const int b = bh >> 4, h = bh & 15;
    const int tid = threadIdx.x, lane = tid & 31, w = tid >> 5;
    __shared__ _Float16 lsV[64 * STR32];   // [kk][t]
    __shared__ _Float16 lsK[64 * STR32];   // [d][t]
    const int kk0 = (w & 1) * 32;
    const int d0  = (w >> 1) * 16;
    v8f acc[2] = {};
    const size_t rowbase = ((size_t)b * LMHA_T) * LMHA_D + (size_t)h * LMHA_DK;

    for (int t0 = 0; t0 < LMHA_T; t0 += 32) {
        // transpose-stage: thread -> t-pair (tid/16) x 4 channels ((tid%16)*4)
        const int ttp = tid >> 4;
        const int g4  = (tid & 15) * 4;
        const size_t r0 = rowbase + (size_t)(t0 + 2 * ttp) * LMHA_D + g4;
        const v4h va = *(const v4h*)(Vf + r0);
        const v4h vb = *(const v4h*)(Vf + r0 + LMHA_D);
        const v4h ka = *(const v4h*)(Kf + r0);
        const v4h kb = *(const v4h*)(Kf + r0 + LMHA_D);
#pragma unroll
        for (int i = 0; i < 4; ++i) {
            v2h pv = { va[i], vb[i] };
            v2h pk = { ka[i], kb[i] };
            *(v2h*)(lsV + (g4 + i) * STR32 + 2 * ttp) = pv;
            *(v2h*)(lsK + (g4 + i) * STR32 + 2 * ttp) = pk;
        }
        __syncthreads();
        v16h af0 = frag_a(lsV, STR32, kk0, lane);
        v16h af1 = frag_a(lsV, STR32, kk0 + 16, lane);
        v16h bf  = frag_b(lsK, STR32, d0, lane);
        acc[0] = wmma_f16(af0, bf, acc[0]);
        acc[1] = wmma_f16(af1, bf, acc[1]);
        __syncthreads();
    }

    const int n = d0 + (lane & 15);
#pragma unroll
    for (int mi = 0; mi < 2; ++mi)
#pragma unroll
        for (int r = 0; r < 8; ++r) {
            const int m = kk0 + mi * 16 + r + ((lane & 16) ? 8 : 0);
            kv[((size_t)bh * 64 + m) * 64 + n] = acc[mi][r];
        }
}

// ---------------------------------------------------------------------------
// attn[b,t,h,kk] = (sum_d q[t,d]*kv[kk,d]) / (q.ksum + eps), f16 out.
// ---------------------------------------------------------------------------
__global__ __launch_bounds__(256) void attn_kernel(
    const _Float16* __restrict__ Qf, const float* __restrict__ kv,
    const float* __restrict__ ksum, _Float16* __restrict__ attn)
{
    const int bh = blockIdx.y;
    const int b = bh >> 4, h = bh & 15;
    const int t0 = blockIdx.x * 128;
    const int tid = threadIdx.x, lane = tid & 31, w = tid >> 5;
    const int wm0 = (w & 3) * 32;
    const int wn0 = (w >> 2) * 32;

    __shared__ _Float16 lsQ[128 * STR64];   // [t][d]
    __shared__ _Float16 lsKV[64 * STR64];   // [kk][d]
    __shared__ float lsS[64];
    __shared__ float lsInv[128];

    const size_t qbase = ((size_t)b * LMHA_T + t0) * LMHA_D + (size_t)h * LMHA_DK;
#pragma unroll
    for (int j = 0; j < 4; ++j) {
        const int flat = tid + j * 256;
        const int row = flat >> 3;
        const int g8  = (flat & 7) * 8;
        cp16(Qf + qbase + (size_t)row * LMHA_D + g8, lsQ + row * STR64 + g8);
    }
#pragma unroll
    for (int j = 0; j < 4; ++j) {
        const int flat = tid + j * 256;
        const int row = flat >> 4;
        const int c4  = (flat & 15) * 4;
        const float4 v = *(const float4*)(kv + (size_t)bh * 4096 + row * 64 + c4);
        v4h hv = { (_Float16)v.x, (_Float16)v.y, (_Float16)v.z, (_Float16)v.w };
        *(v4h*)(lsKV + row * STR64 + c4) = hv;
    }
    if (tid < 64) lsS[tid] = ksum[bh * 64 + tid];
    wait_async();
    __syncthreads();

    if (tid < 128) {
        float s = 0.f;
#pragma unroll 8
        for (int d = 0; d < 64; ++d) s += (float)lsQ[tid * STR64 + d] * lsS[d];
        lsInv[tid] = 1.f / (s + LMHA_EPS);
    }
    __syncthreads();

    v8f acc[2][2] = {};
#pragma unroll
    for (int ks = 0; ks < 2; ++ks) {
        v16h af[2], bf[2];
#pragma unroll
        for (int i = 0; i < 2; ++i) af[i] = frag_a(lsQ + ks * 32, STR64, wm0 + i * 16, lane);
#pragma unroll
        for (int i = 0; i < 2; ++i) bf[i] = frag_b(lsKV + ks * 32, STR64, wn0 + i * 16, lane);
#pragma unroll
        for (int mi = 0; mi < 2; ++mi)
#pragma unroll
            for (int ni = 0; ni < 2; ++ni)
                acc[mi][ni] = wmma_f16(af[mi], bf[ni], acc[mi][ni]);
    }

#pragma unroll
    for (int ni = 0; ni < 2; ++ni) {
        const int kk = wn0 + ni * 16 + (lane & 15);
#pragma unroll
        for (int mi = 0; mi < 2; ++mi)
#pragma unroll
            for (int r = 0; r < 8; ++r) {
                const int tl = wm0 + mi * 16 + r + ((lane & 16) ? 8 : 0);
                const float v = acc[mi][ni][r] * lsInv[tl];
                attn[((size_t)b * LMHA_T + t0 + tl) * LMHA_D + h * LMHA_DV + kk] =
                    (_Float16)v;
            }
    }
}

// ---------------------------------------------------------------------------
extern "C" void kernel_launch(void* const* d_in, const int* in_sizes, int n_in,
                              void* d_out, int out_size, void* d_ws, size_t ws_size,
                              hipStream_t stream)
{
    (void)in_sizes; (void)n_in; (void)out_size; (void)ws_size;
    const float* query = (const float*)d_in[0];
    const float* key   = (const float*)d_in[1];
    const float* value = (const float*)d_in[2];
    const float* Wq = (const float*)d_in[3];
    const float* bq = (const float*)d_in[4];
    const float* Wk = (const float*)d_in[5];
    const float* bk = (const float*)d_in[6];
    const float* Wv = (const float*)d_in[7];
    const float* bv = (const float*)d_in[8];
    const float* Wo = (const float*)d_in[9];
    const float* bo = (const float*)d_in[10];
    float* out = (float*)d_out;

    constexpr size_t NE = (size_t)LMHA_BT * LMHA_D;            // 16M elements
    constexpr size_t KV_BYTES   = (size_t)LMHA_B * LMHA_H * 64 * 64 * 4;
    constexpr size_t KSUM_BYTES = (size_t)LMHA_B * LMHA_H * 64 * 4;
    constexpr size_t W_ELEMS    = (size_t)LMHA_D * LMHA_D;

    char* ws = (char*)d_ws;
    _Float16* Qf = (_Float16*)(ws);
    _Float16* Kf = (_Float16*)(ws + 2 * NE);
    _Float16* Vf = (_Float16*)(ws + 4 * NE);
    _Float16* Af = (_Float16*)(ws + 6 * NE);
    float* kvp   = (float*)(ws + 8 * NE);
    float* ksp   = (float*)(ws + 8 * NE + KV_BYTES);
    _Float16* Wtq = (_Float16*)(ws + 8 * NE + KV_BYTES + KSUM_BYTES);
    _Float16* Wtk = Wtq + W_ELEMS;
    _Float16* Wtv = Wtk + W_ELEMS;
    _Float16* Wto = Wtv + W_ELEMS;

    const dim3 gw(LMHA_D / 64, LMHA_D / 64);
    wconv_kernel<<<gw, 256, 0, stream>>>(Wq, Wtq);
    wconv_kernel<<<gw, 256, 0, stream>>>(Wk, Wtk);
    wconv_kernel<<<gw, 256, 0, stream>>>(Wv, Wtv);
    wconv_kernel<<<gw, 256, 0, stream>>>(Wo, Wto);

    const dim3 gproj(LMHA_BT / 128, LMHA_D / 128);
    gemm_kernel<false, true,  true ><<<gproj, 256, 0, stream>>>(query, Wtq, bq, Qf, LMHA_BT, LMHA_D, LMHA_D);
    gemm_kernel<false, true,  true ><<<gproj, 256, 0, stream>>>(key,   Wtk, bk, Kf, LMHA_BT, LMHA_D, LMHA_D);
    gemm_kernel<false, false, true ><<<gproj, 256, 0, stream>>>(value, Wtv, bv, Vf, LMHA_BT, LMHA_D, LMHA_D);

    ksum_kernel<<<LMHA_B * LMHA_H, 256, 0, stream>>>(Kf, ksp);
    kv_kernel  <<<LMHA_B * LMHA_H, 256, 0, stream>>>(Kf, Vf, kvp);

    attn_kernel<<<dim3(LMHA_T / 128, LMHA_B * LMHA_H), 256, 0, stream>>>(Qf, kvp, ksp, Af);

    gemm_kernel<true, false, false><<<gproj, 256, 0, stream>>>(Af, Wto, bo, out, LMHA_BT, LMHA_D, LMHA_D);
}
